// Block_59322088293038
// MI455X (gfx1250) — matrix operand order repaired and checked by reading, not compiled, not run on hardware
//
#include <hip/hip_runtime.h>

// ---------------------------------------------------------------------------
// GPT block forward for MI455X (gfx1250, wave32, WMMA bf16 -> f32 accum)
// B=2 T=2048 C=768 NH=12 HD=64
// ---------------------------------------------------------------------------

typedef __attribute__((ext_vector_type(16))) __bf16 v16bf;
typedef __attribute__((ext_vector_type(8)))  float  v8f;

struct U32x8 { uint4 a, b; };  // 32 bytes == v16bf

#define kB   2
#define kT   2048
#define kC   768
#define kNH  12
#define kHD  64
#define kM   (kB * kT)        // 4096 rows of activations
#define kQKV (3 * kC)         // 2304

__device__ __forceinline__ v16bf make_frag(const __bf16* p0, const __bf16* p1) {
  U32x8 u;
  u.a = *(const uint4*)p0;
  u.b = *(const uint4*)p1;
  return __builtin_bit_cast(v16bf, u);
}

// ------------------------------ utilities ----------------------------------

__global__ void f32_to_bf16_kernel(const float* __restrict__ src,
                                   __bf16* __restrict__ dst, int n) {
  int i = blockIdx.x * blockDim.x + threadIdx.x;
  if (i < n) dst[i] = (__bf16)src[i];
}

// LayerNorm over C=768, one block (256 thr = 8 waves) per row, bf16 output.
__global__ __launch_bounds__(256) void ln_kernel(
    const float* __restrict__ x, const float* __restrict__ scale,
    const float* __restrict__ bias, __bf16* __restrict__ out) {
  __shared__ float part[8];
  const int row = blockIdx.x;
  const int tid = threadIdx.x;
  const int lane = tid & 31, wave = tid >> 5;
  const float* xr = x + (size_t)row * kC;

  float v[3];
  float s = 0.f;
#pragma unroll
  for (int i = 0; i < 3; i++) { v[i] = xr[tid + i * 256]; s += v[i]; }
#pragma unroll
  for (int off = 16; off >= 1; off >>= 1) s += __shfl_xor(s, off, 32);
  if (lane == 0) part[wave] = s;
  __syncthreads();
  float tot = 0.f;
#pragma unroll
  for (int i = 0; i < 8; i++) tot += part[i];
  const float mu = tot * (1.0f / kC);
  __syncthreads();

  float s2 = 0.f;
#pragma unroll
  for (int i = 0; i < 3; i++) { float d = v[i] - mu; s2 += d * d; }
#pragma unroll
  for (int off = 16; off >= 1; off >>= 1) s2 += __shfl_xor(s2, off, 32);
  if (lane == 0) part[wave] = s2;
  __syncthreads();
  float tot2 = 0.f;
#pragma unroll
  for (int i = 0; i < 8; i++) tot2 += part[i];
  const float rstd = rsqrtf(tot2 * (1.0f / kC) + 1e-5f);

#pragma unroll
  for (int i = 0; i < 3; i++) {
    const int col = tid + i * 256;
    out[(size_t)row * kC + col] =
        (__bf16)((v[i] - mu) * rstd * scale[col] + bias[col]);
  }
}

// ------------------------------ WMMA GEMM ----------------------------------
// C[M,N] = A[M,K] * B[K,N] (+bias), A/B bf16 row-major, f32 accumulate.
// Block tile 128x128, K-step 32. 256 threads = 8 waves, wave tile 64x32.
// A staged row-major (padded stride 40); B staged TRANSPOSED (col-major,
// padded stride 40) so every fragment is two aligned ds_load_b128.

#define BM 128
#define BN 128
#define BK 32
#define LDA 40    // padded LDS stride (elements) for 128x32 A tile
#define LDT 40    // padded LDS stride (elements) for transposed 128x32 B tile

#define EPI_BF16  0   // out_bf = acc + bias
#define EPI_RESID 1   // out_f  = acc + bias + res
#define EPI_GELU  2   // out_bf = gelu(acc + bias)

template <int EPI>
__global__ __launch_bounds__(256) void gemm_bf16_kernel(
    const __bf16* __restrict__ A, const __bf16* __restrict__ Bm,
    const float* __restrict__ bias, const float* __restrict__ res,
    float* __restrict__ outF, __bf16* __restrict__ outB,
    int M, int N, int K) {
  __shared__ __bf16 sA[BM * LDA];
  __shared__ __bf16 sBt[BN * LDT];   // [n][k] transposed

  const int tid  = threadIdx.x;
  const int lane = tid & 31;
  const int wave = tid >> 5;      // 0..7
  const int wm   = wave >> 2;     // 2 waves along M
  const int wn   = wave & 3;      // 4 waves along N
  const int l15   = lane & 15;
  const int lhalf = lane >> 4;
  const int a_base0 = lhalf * 8;
  const int b_baseK = lhalf * 8;

  const int blockM = blockIdx.y * BM;
  const int blockN = blockIdx.x * BN;

  const int aRow = tid >> 1, aCol = (tid & 1) * 16;   // 128x32 tile
  const int bRow = tid >> 3, bCol = (tid & 7) * 16;   // 32x128 tile

  v8f acc[4][2];
  const v8f vzero = {0.f, 0.f, 0.f, 0.f, 0.f, 0.f, 0.f, 0.f};
#pragma unroll
  for (int i = 0; i < 4; i++)
#pragma unroll
    for (int j = 0; j < 2; j++) acc[i][j] = vzero;

  for (int k0 = 0; k0 < K; k0 += BK) {
    const uint4* gA =
        (const uint4*)(A + (size_t)(blockM + aRow) * K + k0 + aCol);
    uint4 va0 = gA[0], va1 = gA[1];
    const uint4* gB =
        (const uint4*)(Bm + (size_t)(k0 + bRow) * N + blockN + bCol);
    uint4 vb0 = gB[0], vb1 = gB[1];

    __syncthreads();  // previous iteration's LDS reads complete
    *(uint4*)&sA[aRow * LDA + aCol]     = va0;
    *(uint4*)&sA[aRow * LDA + aCol + 8] = va1;
    // transpose-scatter B: element j of the loaded row -> column bCol+j
    {
      const unsigned int wds[8] = {vb0.x, vb0.y, vb0.z, vb0.w,
                                   vb1.x, vb1.y, vb1.z, vb1.w};
#pragma unroll
      for (int j = 0; j < 8; j++) {
        const unsigned int w = wds[j];
        sBt[(bCol + 2 * j)     * LDT + bRow] =
            __builtin_bit_cast(__bf16, (unsigned short)(w & 0xffffu));
        sBt[(bCol + 2 * j + 1) * LDT + bRow] =
            __builtin_bit_cast(__bf16, (unsigned short)(w >> 16));
      }
    }
    __syncthreads();

    // A fragments: elem i -> K = base0 + i (i<8), base0 + 8 + i (i>=8)
    v16bf afr[4];
#pragma unroll
    for (int tm = 0; tm < 4; tm++) {
      const __bf16* ap = &sA[(wm * 64 + tm * 16 + l15) * LDA];
      afr[tm] = make_frag(ap + a_base0, ap + a_base0 + 16);
    }
    // B fragments: lane = col, elem i -> K = baseK + i (i<8), baseK + 8 + i (i>=8)
    v16bf bfr[2];
#pragma unroll
    for (int tn = 0; tn < 2; tn++) {
      const __bf16* bp = &sBt[(wn * 32 + tn * 16 + l15) * LDT + b_baseK];
      bfr[tn] = make_frag(bp, bp + 16);
    }
#pragma unroll
    for (int tm = 0; tm < 4; tm++)
#pragma unroll
      for (int tn = 0; tn < 2; tn++)
        acc[tm][tn] = __builtin_amdgcn_wmma_f32_16x16x32_bf16(
            false, afr[tm], false, bfr[tn], (short)0, acc[tm][tn], false, false);
  }

  // Epilogue. C layout: elem r -> M = lhalf*8 + r, N = l15.
  const int halfOff = lhalf * 8;
#pragma unroll
  for (int tm = 0; tm < 4; tm++)
#pragma unroll
    for (int tn = 0; tn < 2; tn++) {
      const int col = blockN + wn * 32 + tn * 16 + l15;
      const float bv = bias[col];
#pragma unroll
      for (int r = 0; r < 8; r++) {
        const int row = blockM + wm * 64 + tm * 16 + halfOff + r;
        const size_t idx = (size_t)row * N + col;
        float v = acc[tm][tn][r] + bv;
        if constexpr (EPI == EPI_RESID) {
          outF[idx] = v + res[idx];
        } else if constexpr (EPI == EPI_GELU) {
          // 0.5*v*(1+tanh(z)) == v * sigmoid(2z)
          const float z = 0.7978845608f * (v + 0.044715f * v * v * v);
          outB[idx] = (__bf16)(v / (1.f + __expf(-2.f * z)));
        } else {
          outB[idx] = (__bf16)v;
        }
      }
    }
}

// --------------------------- flash attention --------------------------------
// Block: 128 threads = 4 waves; each wave owns 16 q rows (q tile = 64).
// KV streamed in 32-row tiles through LDS; online softmax; causal mask.
// qkv layout: [B*T, 2304] bf16 with q at +0, k at +768, v at +1536 (+h*64).
// K kept row-major (S-frag walks d contiguously); V stored TRANSPOSED
// [d][tk] so the PV B-fragment walks tk contiguously.

__global__ __launch_bounds__(128) void attn_kernel(
    const __bf16* __restrict__ qkv, __bf16* __restrict__ ctx) {
  __shared__ __bf16 sK[32 * 72];        // [tk][d]
  __shared__ __bf16 sVt[kHD * 40];      // [d][tk]
  __shared__ __bf16 sP[4][16 * 40];     // per-wave P tile [m][tk]

  const int tid   = threadIdx.x;
  const int lane  = tid & 31;
  const int wave  = tid >> 5;     // 0..3
  const int l15   = lane & 15;
  const int lhalf = lane >> 4;
  const int h = blockIdx.y;
  const int b = blockIdx.z;
  const int qBase = blockIdx.x * 64;
  const int qRow0 = qBase + wave * 16;

  const size_t rstr = kQKV;  // 2304
  const __bf16* base = qkv + (size_t)b * kT * rstr;

  // Q fragments for two K-steps over HD=64
  v16bf qf[2];
  {
    const __bf16* qp = base + (size_t)(qRow0 + l15) * rstr + h * kHD;
#pragma unroll
    for (int ks = 0; ks < 2; ks++) {
      const __bf16* p = qp + ks * 32 + lhalf * 8;
      qf[ks] = make_frag(p, p + 16);
    }
  }

  float mrow[8], lrow[8];
#pragma unroll
  for (int r = 0; r < 8; r++) { mrow[r] = -1e30f; lrow[r] = 0.f; }
  v8f accO[4];
  const v8f vzero = {0.f, 0.f, 0.f, 0.f, 0.f, 0.f, 0.f, 0.f};
#pragma unroll
  for (int i = 0; i < 4; i++) accO[i] = vzero;

  const float scale = 0.125f;  // 1/sqrt(64)
  const int nTiles = (qBase + 64) / 32;
  const int kRow = tid >> 2;            // 0..31 (kv row in tile)
  const int kCol = (tid & 3) * 16;      // 0..48 (d)

  for (int kt = 0; kt < nTiles; kt++) {
    const int kvBase = kt * 32;
    {
      const size_t roff = (size_t)(kvBase + kRow) * rstr + h * kHD + kCol;
      const uint4* gk = (const uint4*)(base + roff + kC);
      const uint4* gv = (const uint4*)(base + roff + 2 * kC);
      uint4 kk0 = gk[0], kk1 = gk[1];
      uint4 vv0 = gv[0], vv1 = gv[1];
      __syncthreads();
      *(uint4*)&sK[kRow * 72 + kCol]     = kk0;
      *(uint4*)&sK[kRow * 72 + kCol + 8] = kk1;
      // V transpose-scatter: loaded row t=kRow, cols d=kCol..kCol+15
      const unsigned int wds[8] = {vv0.x, vv0.y, vv0.z, vv0.w,
                                   vv1.x, vv1.y, vv1.z, vv1.w};
#pragma unroll
      for (int j = 0; j < 8; j++) {
        const unsigned int w = wds[j];
        sVt[(kCol + 2 * j)     * 40 + kRow] =
            __builtin_bit_cast(__bf16, (unsigned short)(w & 0xffffu));
        sVt[(kCol + 2 * j + 1) * 40 + kRow] =
            __builtin_bit_cast(__bf16, (unsigned short)(w >> 16));
      }
      __syncthreads();
    }

    // S = Q * K^T  (16 x 32); B-frag lane = kv row, elem i -> d contiguous
    v8f s[2];
    s[0] = vzero; s[1] = vzero;
#pragma unroll
    for (int sub = 0; sub < 2; sub++) {
      const int col = sub * 16 + l15;  // kv row within tile
#pragma unroll
      for (int ks = 0; ks < 2; ks++) {
        const __bf16* kp = &sK[col * 72 + ks * 32 + lhalf * 8];
        s[sub] = __builtin_amdgcn_wmma_f32_16x16x32_bf16(
            false, qf[ks], false, make_frag(kp, kp + 16), (short)0, s[sub],
            false, false);
      }
    }

    // scale + causal mask + row max
    float rmax[8];
#pragma unroll
    for (int r = 0; r < 8; r++) {
      const int qrow = qRow0 + lhalf * 8 + r;
      float v0 = s[0][r] * scale, v1 = s[1][r] * scale;
      if (kvBase + l15 > qrow)      v0 = -1e30f;
      if (kvBase + 16 + l15 > qrow) v1 = -1e30f;
      s[0][r] = v0; s[1][r] = v1;
      rmax[r] = fmaxf(v0, v1);
    }
#pragma unroll
    for (int off = 8; off >= 1; off >>= 1)
#pragma unroll
      for (int r = 0; r < 8; r++)
        rmax[r] = fmaxf(rmax[r], __shfl_xor(rmax[r], off, 32));

    float corr[8], rsum[8];
#pragma unroll
    for (int r = 0; r < 8; r++) {
      const float mnew = fmaxf(mrow[r], rmax[r]);
      corr[r] = __expf(mrow[r] - mnew);
      mrow[r] = mnew;
      const float p0 = __expf(s[0][r] - mnew);
      const float p1 = __expf(s[1][r] - mnew);
      s[0][r] = p0; s[1][r] = p1;
      rsum[r] = p0 + p1;
    }
#pragma unroll
    for (int off = 8; off >= 1; off >>= 1)
#pragma unroll
      for (int r = 0; r < 8; r++) rsum[r] += __shfl_xor(rsum[r], off, 32);
#pragma unroll
    for (int r = 0; r < 8; r++) lrow[r] = lrow[r] * corr[r] + rsum[r];
#pragma unroll
    for (int i = 0; i < 4; i++)
#pragma unroll
      for (int r = 0; r < 8; r++) accO[i][r] *= corr[r];

    // P (C layout) -> LDS -> A fragment re-layout (wave-local)
    __bf16* pw = sP[wave];
#pragma unroll
    for (int sub = 0; sub < 2; sub++)
#pragma unroll
      for (int r = 0; r < 8; r++)
        pw[(lhalf * 8 + r) * 40 + sub * 16 + l15] = (__bf16)s[sub][r];
    __builtin_amdgcn_wave_barrier();
    asm volatile("s_wait_dscnt 0" ::: "memory");

    const __bf16* pr = pw + l15 * 40 + lhalf * 8;
    const v16bf pa = make_frag(pr, pr + 16);

    // accO += P @ V   ([16x32] x [32x64]); B-frag from transposed V
#pragma unroll
    for (int sub = 0; sub < 4; sub++) {
      const __bf16* vp = &sVt[(sub * 16 + l15) * 40 + lhalf * 8];
      accO[sub] = __builtin_amdgcn_wmma_f32_16x16x32_bf16(
          false, pa, false, make_frag(vp, vp + 16), (short)0, accO[sub],
          false, false);
    }
  }

  // O = accO / l  -> ctx[b,t,h*64+d] bf16
#pragma unroll
  for (int sub = 0; sub < 4; sub++) {
    const int d = sub * 16 + l15;
#pragma unroll
    for (int r = 0; r < 8; r++) {
      const int trow = qRow0 + lhalf * 8 + r;
      const float o = accO[sub][r] / lrow[r];
      ctx[((size_t)(b * kT + trow)) * kC + h * kHD + d] = (__bf16)o;
    }
  }
}

// ------------------------------ launcher ------------------------------------

extern "C" void kernel_launch(void* const* d_in, const int* in_sizes, int n_in,
                              void* d_out, int out_size, void* d_ws,
                              size_t ws_size, hipStream_t stream) {
  const float* x       = (const float*)d_in[0];
  // d_in[1] = mask (causal, applied analytically)
  const float* ln1_s   = (const float*)d_in[2];
  const float* ln1_b   = (const float*)d_in[3];
  const float* wqkv    = (const float*)d_in[4];   // [768,2304]
  const float* bqkv    = (const float*)d_in[5];   // [2304]
  const float* aproj_w = (const float*)d_in[6];   // [768,768]
  const float* aproj_b = (const float*)d_in[7];
  const float* ln2_s   = (const float*)d_in[8];
  const float* ln2_b   = (const float*)d_in[9];
  const float* fc_w    = (const float*)d_in[10];  // [768,3072]
  const float* fc_b    = (const float*)d_in[11];
  const float* proj_w  = (const float*)d_in[12];  // [3072,768]
  const float* proj_b  = (const float*)d_in[13];
  float* out = (float*)d_out;

  char* ws = (char*)d_ws;
  size_t off = 0;
  auto alloc = [&](size_t bytes) {
    char* p = ws + off;
    off = (off + bytes + 255) & ~(size_t)255;
    return p;
  };
  __bf16* h_bf      = (__bf16*)alloc((size_t)kM * kC * 2);
  __bf16* wqkv_bf   = (__bf16*)alloc((size_t)kC * kQKV * 2);
  __bf16* aprojw_bf = (__bf16*)alloc((size_t)kC * kC * 2);
  __bf16* fcw_bf    = (__bf16*)alloc((size_t)kC * 4 * kC * 2);
  __bf16* projw_bf  = (__bf16*)alloc((size_t)4 * kC * kC * 2);
  __bf16* qkv_bf    = (__bf16*)alloc((size_t)kM * kQKV * 2);
  __bf16* ctx_bf    = (__bf16*)alloc((size_t)kM * kC * 2);
  float*  x2        = (float*) alloc((size_t)kM * kC * 4);
  __bf16* h2_bf     = (__bf16*)alloc((size_t)kM * kC * 2);
  __bf16* fcact_bf  = (__bf16*)alloc((size_t)kM * 4 * kC * 2);
  (void)ws_size; (void)in_sizes; (void)n_in; (void)out_size;

  auto cvt = [&](const float* s, __bf16* d, int n) {
    f32_to_bf16_kernel<<<(n + 255) / 256, 256, 0, stream>>>(s, d, n);
  };
  cvt(wqkv,    wqkv_bf,   kC * kQKV);
  cvt(aproj_w, aprojw_bf, kC * kC);
  cvt(fc_w,    fcw_bf,    kC * 4 * kC);
  cvt(proj_w,  projw_bf,  4 * kC * kC);

  // ln1
  ln_kernel<<<kM, 256, 0, stream>>>(x, ln1_s, ln1_b, h_bf);

  // qkv = h @ wqkv + bqkv  -> bf16 [4096, 2304]
  gemm_bf16_kernel<EPI_BF16><<<dim3(kQKV / BN, kM / BM), 256, 0, stream>>>(
      h_bf, wqkv_bf, bqkv, nullptr, nullptr, qkv_bf, kM, kQKV, kC);

  // attention -> ctx bf16 [4096, 768]
  attn_kernel<<<dim3(kT / 64, kNH, kB), 128, 0, stream>>>(qkv_bf, ctx_bf);

  // x2 = x + ctx @ attn_proj_w + attn_proj_b
  gemm_bf16_kernel<EPI_RESID><<<dim3(kC / BN, kM / BM), 256, 0, stream>>>(
      ctx_bf, aprojw_bf, aproj_b, x, x2, nullptr, kM, kC, kC);

  // ln2
  ln_kernel<<<kM, 256, 0, stream>>>(x2, ln2_s, ln2_b, h2_bf);

  // fcact = gelu(h2 @ fc_w + fc_b) -> bf16 [4096, 3072]
  gemm_bf16_kernel<EPI_GELU><<<dim3(4 * kC / BN, kM / BM), 256, 0, stream>>>(
      h2_bf, fcw_bf, fc_b, nullptr, nullptr, fcact_bf, kM, 4 * kC, kC);

  // out = x2 + fcact @ proj_w + proj_b
  gemm_bf16_kernel<EPI_RESID><<<dim3(kC / BN, kM / BM), 256, 0, stream>>>(
      fcact_bf, projw_bf, proj_b, x2, out, nullptr, kM, kC, 4 * kC);
}
